// Encoder_block_74878459838600
// MI455X (gfx1250) — matrix operand order repaired and checked by reading, not compiled, not run on hardware
//
#include <hip/hip_runtime.h>

// ---------------------------------------------------------------------------
// Transformer encoder block for MI455X (gfx1250, wave32, WMMA + TDM).
// Compute-bound (~138 GFLOP vs ~120MB working set, L2-resident): all GEMM-
// shaped math on v_wmma_f32_16x16x32_bf16 (bf16 in, f32 accumulate).
// GEMM operands staged by the Tensor Data Mover (tensor_load_to_lds) into
// double-buffered, hardware-padded LDS tiles (K-step 64 per DMA stage);
// attention softmax reductions run on v_permlane16 (VALU-only).
// ---------------------------------------------------------------------------

typedef __bf16 bf16_t;
typedef __attribute__((ext_vector_type(16))) __bf16 v16bf;
typedef __attribute__((ext_vector_type(8)))  __bf16 v8bf;
typedef __attribute__((ext_vector_type(4)))  __bf16 v4bf;
typedef __attribute__((ext_vector_type(8)))  float  v8f;
typedef __attribute__((ext_vector_type(4)))  unsigned int u32x4;
typedef __attribute__((ext_vector_type(8)))  int i32x8;
typedef __attribute__((ext_vector_type(4)))  int i32x4;

constexpr int kB   = 2;
constexpr int kS   = 2048;
constexpr int kD   = 1024;
constexpr int kH   = 16;
constexpr int kHD  = 64;
constexpr int kHID = 4096;
constexpr int kM   = kB * kS;      // 4096 token rows
constexpr float kEPS = 1e-5f;
constexpr float kNEG = -1e9f;

// GEMM LDS tile: rows of 64 bf16 (128B) + 16B TDM pad -> 144B stride.
// 144B is 16B-aligned for b128 ds reads; row*36 dwords mod 64 -> 16 distinct
// bank groups (conflict-free fragment reads).
constexpr int kGLd = 72;           // gemm LDS row stride (elems)
// Attention LDS tiles: rows of 32 bf16 (64B) + 16B pad -> 80B stride.
constexpr int kALd = 40;           // attention LDS row stride (elems)

#if __has_builtin(__builtin_amdgcn_tensor_load_to_lds)
#define HAVE_TDM 1
#else
#warning "gfx1250 tensor_load_to_lds builtin unavailable; using vector-load fallback"
#endif

// ---------------------------------------------------------------------------
// WMMA fragment loaders (layouts per CDNA5 ISA 7.12.2, wave32).
// A (16x32): lane={half,m}; elems K = half*8+0..7 and 16+half*8+0..7.
// B (32x16, W row-major [N,K]): lane={half,n}; elems K = half*8+0..7 and 16+half*8+0..7.
// C/D (16x16 f32): vgpr r holds row m = r + half*8, column n = lane&15.
// ---------------------------------------------------------------------------
union AB_t { v16bf v; v8bf h[2]; };

__device__ inline v16bf load_a_frag(const bf16_t* __restrict__ base, int ld) {
  const int lane = threadIdx.x & 31;
  const int hf = lane >> 4, m = lane & 15;
  const bf16_t* p = base + (size_t)m * ld;
  AB_t u;
  u.h[0] = *(const v8bf*)(p + hf * 8);
  u.h[1] = *(const v8bf*)(p + 16 + hf * 8);
  return u.v;
}

__device__ inline v16bf load_b_frag(const bf16_t* __restrict__ base, int ld) {
  const int lane = threadIdx.x & 31;
  const int hf = lane >> 4, n = lane & 15;
  const bf16_t* p = base + (size_t)n * ld;
  AB_t u;
  u.h[0] = *(const v8bf*)(p + hf * 8);
  u.h[1] = *(const v8bf*)(p + 16 + hf * 8);
  return u.v;
}

__device__ inline v8f wmma_bf16(v16bf a, v16bf b, v8f c) {
  return __builtin_amdgcn_wmma_f32_16x16x32_bf16(false, a, false, b,
                                                 (short)0, c, false, false);
}

// ---------------------------------------------------------------------------
// Cross-lane reductions within 16-lane halves: v_permlane16_b32 (VALU only).
// sel nibbles implement lane ^= {1,2,4,8}.
// ---------------------------------------------------------------------------
#if __has_builtin(__builtin_amdgcn_permlane16)
__device__ inline float permx16(float v, unsigned lo, unsigned hi) {
  int i = __float_as_int(v);
  int r = __builtin_amdgcn_permlane16(i, i, (int)lo, (int)hi, false, false);
  return __int_as_float(r);
}
__device__ inline float red_max16(float v) {
  v = fmaxf(v, permx16(v, 0x67452301u, 0xEFCDAB89u));  // xor 1
  v = fmaxf(v, permx16(v, 0x54761032u, 0xDCFE98BAu));  // xor 2
  v = fmaxf(v, permx16(v, 0x32107654u, 0xBA98FEDCu));  // xor 4
  v = fmaxf(v, permx16(v, 0xFEDCBA98u, 0x76543210u));  // xor 8
  return v;
}
__device__ inline float red_sum16(float v) {
  v += permx16(v, 0x67452301u, 0xEFCDAB89u);
  v += permx16(v, 0x54761032u, 0xDCFE98BAu);
  v += permx16(v, 0x32107654u, 0xBA98FEDCu);
  v += permx16(v, 0xFEDCBA98u, 0x76543210u);
  return v;
}
#else
__device__ inline float red_max16(float v) {
  for (int off = 8; off > 0; off >>= 1) v = fmaxf(v, __shfl_xor(v, off, 32));
  return v;
}
__device__ inline float red_sum16(float v) {
  for (int off = 8; off > 0; off >>= 1) v += __shfl_xor(v, off, 32);
  return v;
}
#endif

// ---------------------------------------------------------------------------
// Tensor Data Mover: 2D tile load (bf16) global -> LDS, rows of 64 elems
// (128B) with 16B pad after each row (pad_interval=4: 32 dwords,
// pad_amount=3: 4 dwords). Descriptor packing per CDNA5 ISA 8.3/8.4.
// Issued by one wave; completion via TENSORcnt.
// ---------------------------------------------------------------------------
#ifdef HAVE_TDM
__device__ inline void tdm_load_tile(unsigned lds_addr, const bf16_t* gptr,
                                     unsigned rows, unsigned tensor_rows,
                                     unsigned row_stride_elems) {
  const unsigned long long ga = (unsigned long long)(const void*)gptr;
  u32x4 g0;
  g0.x = 1u;                                   // count=1, user mode
  g0.y = lds_addr;                             // LDS byte address
  g0.z = (unsigned)(ga & 0xFFFFFFFFu);         // global_addr[31:0]
  g0.w = (unsigned)((ga >> 32) & 0x01FFFFFFu)  // global_addr[56:32]
       | (2u << 30);                           // type = 2 (image)
  const unsigned tk = row_stride_elems;        // tensor_dim0 (>= tile cols)
  const unsigned tr = tensor_rows;             // tensor_dim1
  i32x8 g1;
  g1[0] = (int)((1u << 16)                     // data_size = 2 bytes
              | (1u << 20)                     // pad_enable
              | (4u << 22)                     // pad_interval: 32 dwords
              | (3u << 25));                   // pad_amount: 4 dwords (16B)
  g1[1] = (int)((tk & 0xFFFFu) << 16);         // tensor_dim0 lo16
  g1[2] = (int)((tk >> 16) | ((tr & 0xFFFFu) << 16)); // dim0 hi / dim1 lo
  g1[3] = (int)((tr >> 16) | (64u << 16));     // dim1 hi / tile_dim0 = 64
  g1[4] = (int)rows;                           // tile_dim1, tile_dim2 = 0
  g1[5] = (int)row_stride_elems;               // tensor_dim0_stride lo32
  g1[6] = 0;                                   // stride hi16 / dim1_stride lo
  g1[7] = 0;
  i32x4 z4 = {};
#if defined(__clang_major__) && __clang_major__ >= 23
  i32x8 z8 = {};
  __builtin_amdgcn_tensor_load_to_lds(g0, g1, z4, z4, z8, 0);
#else
  __builtin_amdgcn_tensor_load_to_lds(g0, g1, z4, z4, 0);
#endif
}
#endif

// ---------------------------------------------------------------------------
// fp32 -> bf16 converts (weights), plain and transposing.
// ---------------------------------------------------------------------------
__global__ __launch_bounds__(256) void cvt_bf16(const float* __restrict__ in,
                                                bf16_t* __restrict__ out, int n) {
  int i = blockIdx.x * 256 + threadIdx.x;
  if (i < n) out[i] = (bf16_t)in[i];
}

__global__ __launch_bounds__(256) void cvt_t_bf16(const float* __restrict__ in,
                                                  bf16_t* __restrict__ out,
                                                  int rows, int cols) {
  int i = blockIdx.x * 256 + threadIdx.x;
  if (i < rows * cols) {
    int r = i / cols, c = i % cols;
    out[(size_t)c * rows + r] = (bf16_t)in[i];  // out is [cols][rows]
  }
}

// ---------------------------------------------------------------------------
// LayerNorm: one 256-thread block per row, f32 math, bf16 out.
// ---------------------------------------------------------------------------
__global__ __launch_bounds__(256) void ln_kernel(const float* __restrict__ x,
                                                 const float* __restrict__ g,
                                                 const float* __restrict__ b,
                                                 bf16_t* __restrict__ out, int n) {
  const int row = blockIdx.x;
  const float* xr = x + (size_t)row * n;
  float s = 0.f, s2 = 0.f;
  for (int i = threadIdx.x; i < n; i += 256) {
    float v = xr[i];
    s += v; s2 += v * v;
  }
#pragma unroll
  for (int off = 16; off > 0; off >>= 1) {
    s  += __shfl_xor(s,  off, 32);
    s2 += __shfl_xor(s2, off, 32);
  }
  __shared__ float sh0[8], sh1[8];
  const int w = threadIdx.x >> 5;
  if ((threadIdx.x & 31) == 0) { sh0[w] = s; sh1[w] = s2; }
  __syncthreads();
  if (threadIdx.x < 32) {
    s  = (threadIdx.x < 8) ? sh0[threadIdx.x] : 0.f;
    s2 = (threadIdx.x < 8) ? sh1[threadIdx.x] : 0.f;
#pragma unroll
    for (int off = 4; off > 0; off >>= 1) {
      s  += __shfl_xor(s,  off, 32);
      s2 += __shfl_xor(s2, off, 32);
    }
    if (threadIdx.x == 0) { sh0[0] = s; sh1[0] = s2; }
  }
  __syncthreads();
  const float mean = sh0[0] / n;
  const float var  = sh1[0] / n - mean * mean;
  const float rstd = rsqrtf(var + kEPS);
  for (int i = threadIdx.x; i < n; i += 256)
    out[(size_t)row * n + i] = (bf16_t)((xr[i] - mean) * rstd * g[i] + b[i]);
}

// ---------------------------------------------------------------------------
// WMMA GEMM:  C[M,N] = act( A[M,K] @ W[N,K]^T + bias ) + resid
// 256-thread block = 8 waves computes a 128x128 tile; waves in a 2(M)x4(N)
// grid each own 64x32 (8 C frags). Per pipeline stage (K-step 64): TDM DMAs
// A/B 128x64 tiles into padded LDS (double-buffered, s_wait_tensorcnt +
// barrier), then two 32-K substeps of 6 fragment ds reads + 8 WMMAs/wave.
// ---------------------------------------------------------------------------
template <bool BIAS, bool RELU, bool RESID, bool OUTB>
__global__ __launch_bounds__(256) void gemm_wmma(const bf16_t* __restrict__ A,
                                                 const bf16_t* __restrict__ W,
                                                 const float* __restrict__ bias,
                                                 const float* __restrict__ resid,
                                                 float* __restrict__ outF,
                                                 bf16_t* __restrict__ outB,
                                                 int M, int N, int K) {
  const int n0 = blockIdx.x * 128;
  const int m0 = blockIdx.y * 128;
  const int wid = threadIdx.x >> 5;
  const int wm = wid >> 2;            // 0..1  -> 64 M-rows each
  const int wn = wid & 3;             // 0..3  -> 32 N-cols each

  __shared__ bf16_t tA[2][128][kGLd];
  __shared__ bf16_t tB[2][128][kGLd];

  v8f c[4][2] = {};

#ifdef HAVE_TDM
  if (wid == 0) {
    tdm_load_tile((unsigned)(size_t)&tA[0][0][0], A + (size_t)m0 * K, 128, M, K);
    tdm_load_tile((unsigned)(size_t)&tB[0][0][0], W + (size_t)n0 * K, 128, N, K);
  }
  int buf = 0;
  for (int k = 0; k < K; k += 64) {
    if (wid == 0) __builtin_amdgcn_s_wait_tensorcnt(0);
    __syncthreads();                  // tiles visible; prev-buf readers done
    if (wid == 0 && k + 64 < K) {
      tdm_load_tile((unsigned)(size_t)&tA[buf ^ 1][0][0],
                    A + (size_t)m0 * K + (k + 64), 128, M, K);
      tdm_load_tile((unsigned)(size_t)&tB[buf ^ 1][0][0],
                    W + (size_t)n0 * K + (k + 64), 128, N, K);
    }
#pragma unroll
    for (int s = 0; s < 2; s++) {     // two 32-K substeps per DMA stage
      v16bf a[4], b[2];
#pragma unroll
      for (int i = 0; i < 4; i++)
        a[i] = load_a_frag(&tA[buf][wm * 64 + 16 * i][s * 32], kGLd);
#pragma unroll
      for (int j = 0; j < 2; j++)
        b[j] = load_b_frag(&tB[buf][wn * 32 + 16 * j][s * 32], kGLd);
#pragma unroll
      for (int i = 0; i < 4; i++)
#pragma unroll
        for (int j = 0; j < 2; j++)
          c[i][j] = wmma_bf16(a[i], b[j], c[i][j]);
    }
    buf ^= 1;
  }
#else
  for (int k = 0; k < K; k += 64) {
    __syncthreads();
#pragma unroll
    for (int t = 0; t < 4; t++) {     // 1024 16B chunks per tile, 256 thr
      const int cidx = t * 256 + threadIdx.x;
      const int r = cidx >> 3, c8 = (cidx & 7) * 8;
      *(v8bf*)&tA[0][r][c8] = *(const v8bf*)(A + (size_t)(m0 + r) * K + k + c8);
      *(v8bf*)&tB[0][r][c8] = *(const v8bf*)(W + (size_t)(n0 + r) * K + k + c8);
    }
    __syncthreads();
#pragma unroll
    for (int s = 0; s < 2; s++) {
      v16bf a[4], b[2];
#pragma unroll
      for (int i = 0; i < 4; i++)
        a[i] = load_a_frag(&tA[0][wm * 64 + 16 * i][s * 32], kGLd);
#pragma unroll
      for (int j = 0; j < 2; j++)
        b[j] = load_b_frag(&tB[0][wn * 32 + 16 * j][s * 32], kGLd);
#pragma unroll
      for (int i = 0; i < 4; i++)
#pragma unroll
        for (int j = 0; j < 2; j++)
          c[i][j] = wmma_bf16(a[i], b[j], c[i][j]);
    }
  }
#endif

  const int lane = threadIdx.x & 31;
  const int hf = lane >> 4, nl = lane & 15;
#pragma unroll
  for (int i = 0; i < 4; i++)
#pragma unroll
    for (int j = 0; j < 2; j++)
#pragma unroll
      for (int r = 0; r < 8; r++) {
        const int row = m0 + wm * 64 + 16 * i + r + hf * 8;
        const int col = n0 + wn * 32 + 16 * j + nl;
        float v = c[i][j][r];
        if (BIAS) v += bias[col];
        if (RELU) v = fmaxf(v, 0.f);
        if (RESID) v += resid[(size_t)row * N + col];
        if (OUTB) outB[(size_t)row * N + col] = (bf16_t)v;
        else      outF[(size_t)row * N + col] = v;
      }
}

// ---------------------------------------------------------------------------
// Flash attention: one wave per (batch, head, 16-row q-tile).
// Streams K/V in 32-wide tiles with online softmax; scores and P*V on WMMA.
// Row reductions via v_permlane16; P goes C-layout -> A-layout through LDS;
// V staged transposed in LDS (padded rows) for contiguous P@V B-frag reads.
// ---------------------------------------------------------------------------
__global__ __launch_bounds__(32) void attn_kernel(const bf16_t* __restrict__ Q,
                                                  const bf16_t* __restrict__ Kt,
                                                  const bf16_t* __restrict__ Vt,
                                                  const int* __restrict__ mask,
                                                  bf16_t* __restrict__ O) {
  const int q0 = blockIdx.x * 16;
  const int h  = blockIdx.y;
  const int bb = blockIdx.z;
  const int lane = threadIdx.x;
  const int hf = lane >> 4, nl = lane & 15;

  __shared__ bf16_t vT[kHD][kALd];  // V tile transposed: [d][kcol]
  __shared__ bf16_t pT[16][kALd];   // P tile: [q-row][kcol]

  const bf16_t* qbase = Q + ((size_t)bb * kS + q0) * kD + h * kHD;
  v16bf qa0 = load_a_frag(qbase, kD);        // d = 0..31
  v16bf qa1 = load_a_frag(qbase + 32, kD);   // d = 32..63

  v8f o[4] = {};           // O accum over d = 0..63 (4 x 16 cols)
  float mrow[8], lrow[8];
#pragma unroll
  for (int r = 0; r < 8; r++) { mrow[r] = -3.0e38f; lrow[r] = 0.f; }

  for (int k0 = 0; k0 < kS; k0 += 32) {
    // ---- stage V tile (32 x 64) transposed into LDS ---------------------
#pragma unroll
    for (int i = 0; i < 16; i++) {
      const int vidx = i * 32 + lane;          // 512 v4bf chunks
      const int r  = vidx >> 4;                // 16 chunks per row
      const int c4 = (vidx & 15) * 4;
      const bf16_t* src = Vt + ((size_t)bb * kS + k0 + r) * kD + h * kHD + c4;
      v4bf d = *(const v4bf*)src;
#pragma unroll
      for (int t = 0; t < 4; t++) vT[c4 + t][r] = d[t];
    }
    __syncthreads();

    // ---- scores: S = Q K^T / sqrt(HD), two 16x16 tiles ------------------
    v8f s[2] = {};
#pragma unroll
    for (int t = 0; t < 2; t++) {
      const bf16_t* kb = Kt + ((size_t)bb * kS + k0 + t * 16) * kD + h * kHD;
      v16bf b0 = load_b_frag(kb, kD);
      v16bf b1 = load_b_frag(kb + 32, kD);
      s[t] = wmma_bf16(qa0, b0, s[t]);
      s[t] = wmma_bf16(qa1, b1, s[t]);
    }

    float sc[2][8];
#pragma unroll
    for (int t = 0; t < 2; t++) {
      const int kcol = k0 + t * 16 + nl;
      const float mvalid = (mask[bb * kS + kcol] == 0) ? kNEG : 0.f;
#pragma unroll
      for (int r = 0; r < 8; r++) sc[t][r] = s[t][r] * 0.125f + mvalid;
    }

    // ---- online softmax update ------------------------------------------
    float alpha[8], mnew[8];
#pragma unroll
    for (int r = 0; r < 8; r++) {
      float mx = red_max16(fmaxf(sc[0][r], sc[1][r]));
      mnew[r]  = fmaxf(mrow[r], mx);
      alpha[r] = __expf(mrow[r] - mnew[r]);
      mrow[r]  = mnew[r];
    }
#pragma unroll
    for (int t = 0; t < 2; t++)
#pragma unroll
      for (int r = 0; r < 8; r++) sc[t][r] = __expf(sc[t][r] - mnew[r]);
#pragma unroll
    for (int r = 0; r < 8; r++) {
      float rs = red_sum16(sc[0][r] + sc[1][r]);
      lrow[r] = lrow[r] * alpha[r] + rs;
    }

    // ---- write P tile to LDS (C layout -> memory), rescale O ------------
#pragma unroll
    for (int t = 0; t < 2; t++)
#pragma unroll
      for (int r = 0; r < 8; r++) pT[r + hf * 8][t * 16 + nl] = (bf16_t)sc[t][r];
#pragma unroll
    for (int f = 0; f < 4; f++)
#pragma unroll
      for (int r = 0; r < 8; r++) o[f][r] *= alpha[r];
    __syncthreads();

    // ---- O += P @ V : P as A-frag (16x32), V^T rows as B-frags ----------
    AB_t pa;
    pa.h[0] = *(const v8bf*)&pT[nl][hf * 8];
    pa.h[1] = *(const v8bf*)&pT[nl][16 + hf * 8];
#pragma unroll
    for (int f = 0; f < 4; f++) {
      AB_t vb;
      vb.h[0] = *(const v8bf*)&vT[f * 16 + nl][hf * 8];
      vb.h[1] = *(const v8bf*)&vT[f * 16 + nl][16 + hf * 8];
      o[f] = wmma_bf16(pa.v, vb.v, o[f]);
    }
    __syncthreads();
  }

  // ---- normalize + store (bf16, heads concatenated along D) --------------
#pragma unroll
  for (int r = 0; r < 8; r++) {
    const float inv = 1.0f / fmaxf(lrow[r], 1e-20f);
    const size_t row = (size_t)bb * kS + q0 + r + hf * 8;
#pragma unroll
    for (int f = 0; f < 4; f++)
      O[row * kD + h * kHD + f * 16 + nl] = (bf16_t)(o[f][r] * inv);
  }
}

// ---------------------------------------------------------------------------
// Host-side launch.
// ---------------------------------------------------------------------------
extern "C" void kernel_launch(void* const* d_in, const int* in_sizes, int n_in,
                              void* d_out, int out_size, void* d_ws, size_t ws_size,
                              hipStream_t stream) {
  (void)in_sizes; (void)n_in; (void)out_size; (void)ws_size;
  const float* x     = (const float*)d_in[0];
  const int*   mask  = (const int*)d_in[1];
  const float* w_q   = (const float*)d_in[2];
  const float* w_k   = (const float*)d_in[3];
  const float* w_v   = (const float*)d_in[4];
  const float* w_o   = (const float*)d_in[5];
  const float* ln1_g = (const float*)d_in[6];
  const float* ln1_b = (const float*)d_in[7];
  const float* ln2_g = (const float*)d_in[8];
  const float* ln2_b = (const float*)d_in[9];
  const float* w1    = (const float*)d_in[10];
  const float* b1    = (const float*)d_in[11];
  const float* w2    = (const float*)d_in[12];
  const float* b2    = (const float*)d_in[13];
  float* out = (float*)d_out;

  char* ws = (char*)d_ws;
  const size_t MB = 1ull << 20;
  bf16_t* wq_b  = (bf16_t*)(ws + 0 * MB);    // 2 MB
  bf16_t* wk_b  = (bf16_t*)(ws + 2 * MB);    // 2 MB
  bf16_t* wv_b  = (bf16_t*)(ws + 4 * MB);    // 2 MB
  bf16_t* wo_b  = (bf16_t*)(ws + 6 * MB);    // 2 MB
  bf16_t* w1t_b = (bf16_t*)(ws + 8 * MB);    // 8 MB  [HID][D]
  bf16_t* w2t_b = (bf16_t*)(ws + 16 * MB);   // 8 MB  [D][HID]
  bf16_t* xn_b  = (bf16_t*)(ws + 24 * MB);   // 8 MB  LN1 out
  bf16_t* q_b   = (bf16_t*)(ws + 32 * MB);   // 8 MB
  bf16_t* k_b   = (bf16_t*)(ws + 40 * MB);   // 8 MB
  bf16_t* v_b   = (bf16_t*)(ws + 48 * MB);   // 8 MB
  bf16_t* at_b  = (bf16_t*)(ws + 56 * MB);   // 8 MB  attention out
  float*  x1    = (float*) (ws + 64 * MB);   // 16 MB residual-1
  bf16_t* h_b   = (bf16_t*)(ws + 80 * MB);   // 8 MB  LN2 out
  bf16_t* ff1_b = (bf16_t*)(ws + 88 * MB);   // 32 MB relu(h@w1+b1)

  // weight converts
  cvt_bf16<<<(kD * kD) / 256, 256, 0, stream>>>(w_q, wq_b, kD * kD);
  cvt_bf16<<<(kD * kD) / 256, 256, 0, stream>>>(w_k, wk_b, kD * kD);
  cvt_bf16<<<(kD * kD) / 256, 256, 0, stream>>>(w_v, wv_b, kD * kD);
  cvt_bf16<<<(kD * kD) / 256, 256, 0, stream>>>(w_o, wo_b, kD * kD);
  cvt_t_bf16<<<(kD * kHID) / 256, 256, 0, stream>>>(w1, w1t_b, kD, kHID);
  cvt_t_bf16<<<(kHID * kD) / 256, 256, 0, stream>>>(w2, w2t_b, kHID, kD);

  // LN1
  ln_kernel<<<kM, 256, 0, stream>>>(x, ln1_g, ln1_b, xn_b, kD);

  // QKV projections (bf16 out)
  dim3 gD(kD / 128, kM / 128);
  gemm_wmma<false, false, false, true><<<gD, 256, 0, stream>>>(
      xn_b, wq_b, nullptr, nullptr, nullptr, q_b, kM, kD, kD);
  gemm_wmma<false, false, false, true><<<gD, 256, 0, stream>>>(
      xn_b, wk_b, nullptr, nullptr, nullptr, k_b, kM, kD, kD);
  gemm_wmma<false, false, false, true><<<gD, 256, 0, stream>>>(
      xn_b, wv_b, nullptr, nullptr, nullptr, v_b, kM, kD, kD);

  // attention
  attn_kernel<<<dim3(kS / 16, kH, kB), 32, 0, stream>>>(q_b, k_b, v_b, mask, at_b);

  // x1 = x + attn @ w_o^T   (f32 out)
  gemm_wmma<false, false, true, false><<<gD, 256, 0, stream>>>(
      at_b, wo_b, nullptr, x, x1, nullptr, kM, kD, kD);

  // LN2
  ln_kernel<<<kM, 256, 0, stream>>>(x1, ln2_g, ln2_b, h_b, kD);

  // ff1 = relu(h @ w1 + b1)   (bf16 out)
  dim3 gH(kHID / 128, kM / 128);
  gemm_wmma<true, true, false, true><<<gH, 256, 0, stream>>>(
      h_b, w1t_b, b1, nullptr, nullptr, ff1_b, kM, kHID, kD);

  // out = x1 + ff1 @ w2 + b2   (f32 out)
  gemm_wmma<true, false, true, false><<<gD, 256, 0, stream>>>(
      ff1_b, w2t_b, b2, x1, out, nullptr, kM, kD, kHID);
}
